// SimpleGCNRes_48455821033980
// MI455X (gfx1250) — matrix operand order repaired and checked by reading, not compiled, not run on hardware
//
#include <hip/hip_runtime.h>
#include <hip/hip_bf16.h>

typedef __attribute__((ext_vector_type(16))) _Float16 v16h;
typedef __attribute__((ext_vector_type(8)))  _Float16 v8h;
typedef __attribute__((ext_vector_type(4)))  _Float16 v4h;
typedef __attribute__((ext_vector_type(8)))  float    v8f;

#define GCN_N 50000
#define GCN_E 1600000
#define GCN_FIN 256
#define GCN_H 128
#define GCN_C 40

// ---------------------------------------------------------------------------
// degree / norm kernels
// ---------------------------------------------------------------------------
__global__ void gcn_deg_init(float* __restrict__ deg, int n) {
    int i = blockIdx.x * blockDim.x + threadIdx.x;
    if (i < n) deg[i] = 1.0f;   // self-loop contributes 1
}

__global__ void gcn_deg_edges(const int* __restrict__ dst, float* __restrict__ deg, int e) {
    int i = blockIdx.x * blockDim.x + threadIdx.x;
    if (i < e) atomicAdd(&deg[dst[i]], 1.0f);
}

__global__ void gcn_dinv(float* __restrict__ d, int n) {
    int i = blockIdx.x * blockDim.x + threadIdx.x;
    if (i < n) d[i] = rsqrtf(d[i]);   // deg >= 1 always (self-loop)
}

// ---------------------------------------------------------------------------
// conversion helpers (float4 vectorized)
// ---------------------------------------------------------------------------
__global__ void gcn_f32_to_f16_v4(const float4* __restrict__ in, v4h* __restrict__ out, int n4) {
    int i = blockIdx.x * blockDim.x + threadIdx.x;
    if (i >= n4) return;
    float4 v = in[i];
    v4h h = { (_Float16)v.x, (_Float16)v.y, (_Float16)v.z, (_Float16)v.w };
    out[i] = h;
}

// W is K x OUT (row-major f32); Wt is OUTPAD x K (row-major f16), zero-padded rows.
__global__ void gcn_w_transpose(const float* __restrict__ W, _Float16* __restrict__ Wt,
                                int K, int OUT, int OUTPAD) {
    int idx = blockIdx.x * blockDim.x + threadIdx.x;
    if (idx >= K * OUTPAD) return;
    int o = idx / K;
    int k = idx - o * K;
    Wt[(size_t)o * K + k] = (o < OUT) ? (_Float16)W[(size_t)k * OUT + o] : (_Float16)0.0f;
}

// ---------------------------------------------------------------------------
// WMMA GEMM:  C[M x ldC] = A[M x K](f16) * Bt[OUTPAD x K](f16)^T  (+ bias)
// K is a compile-time constant -> fully unrolled K-loop, immediate-offset
// ds_load_b128 / global_load_b128 clauses, backend-scheduled waits.
// A tile (16 x K, contiguous rows) staged in LDS once per block and shared by
// all waves; one wave per 16x16 output tile; blockDim.x = 32 * tilesN;
// gridDim.x = M/16.
// ---------------------------------------------------------------------------
template <int K>
__global__ void gcn_wmma_gemm(const _Float16* __restrict__ A,
                              const _Float16* __restrict__ Bt,
                              const float* __restrict__ bias,   // may be nullptr
                              float* __restrict__ C,
                              int OUT, int ldC) {
    __shared__ _Float16 lds_a[16 * K];

    const int lane  = threadIdx.x & 31;
    const int wave  = threadIdx.x >> 5;
    const int tileM = blockIdx.x;
    const int tileN = wave;
    const int l15   = lane & 15;

    // --- cooperative stage of the contiguous 16 x K A tile into LDS ---
    {
        const v8h* srcA = (const v8h*)(A + (size_t)tileM * 16 * K);
        v8h* dstA = (v8h*)lds_a;
        const int nvec = (16 * K) / 8;
        for (int i = threadIdx.x; i < nvec; i += blockDim.x) {
            __builtin_prefetch((const void*)(srcA + i + blockDim.x), 0, 3);
            dstA[i] = srcA[i];
        }
    }
    __syncthreads();

    const int colB = tileN * 16 + l15;
    const int aoff = (lane < 16) ? 0 : 8;     // A: K sub-offset for upper half-wave
    const int boff = (lane < 16) ? 0 : 8;     // B: K sub-offset for upper half-wave
    const _Float16* aptr = lds_a + l15 * K + aoff;             // LDS
    const _Float16* bptr = Bt + (size_t)colB * K + boff;       // global (L2-resident)

    v8f acc = {};
#pragma unroll
    for (int k0 = 0; k0 < K; k0 += 32) {
        v8h a_lo = *(const v8h*)(aptr + k0);
        v8h a_hi = *(const v8h*)(aptr + k0 + 16);
        v8h b_lo = *(const v8h*)(bptr + k0);
        v8h b_hi = *(const v8h*)(bptr + k0 + 16);
        v16h a = __builtin_shufflevector(a_lo, a_hi, 0,1,2,3,4,5,6,7,8,9,10,11,12,13,14,15);
        v16h b = __builtin_shufflevector(b_lo, b_hi, 0,1,2,3,4,5,6,7,8,9,10,11,12,13,14,15);
        acc = __builtin_amdgcn_wmma_f32_16x16x32_f16(
                  /*neg_a=*/false, a, /*neg_b=*/false, b,
                  /*c_mod=*/(short)0, acc, /*reuse_a=*/false, /*reuse_b=*/false);
    }

    const int n = colB;
    if (n < OUT) {
        float bv = bias ? bias[n] : 0.0f;
        int m0 = tileM * 16 + ((lane < 16) ? 0 : 8);
#pragma unroll
        for (int j = 0; j < 8; ++j)
            C[(size_t)(m0 + j) * ldC + n] = acc[j] + bv;
    }
}

// ---------------------------------------------------------------------------
// aggregation: self-loop init + bias, then edge scatter with f32 atomics
// ---------------------------------------------------------------------------
__global__ void gcn_agg_init(const float4* __restrict__ XW4, const float* __restrict__ dinv,
                             const float4* __restrict__ bias4, float4* __restrict__ AGG4,
                             int n, int OUT4) {
    int idx = blockIdx.x * blockDim.x + threadIdx.x;
    if (idx >= n * OUT4) return;
    int i = idx / OUT4;
    int f = idx - i * OUT4;
    float dv = dinv[i];
    float s = dv * dv;
    float4 v = XW4[idx];
    float4 b = bias4[f];
    float4 r = { s * v.x + b.x, s * v.y + b.y, s * v.z + b.z, s * v.w + b.w };
    AGG4[idx] = r;
}

// one wave per edge; lanes stride over float4 chunks (coalesced within a row)
__global__ void gcn_edge_scatter(const float4* __restrict__ XW4, const float* __restrict__ dinv,
                                 const int* __restrict__ src, const int* __restrict__ dst,
                                 float* __restrict__ AGG, int e, int OUT4) {
    int gid  = blockIdx.x * blockDim.x + threadIdx.x;
    int edge = gid >> 5;
    int lane = gid & 31;
    if (edge >= e) return;
    int s = src[edge];
    int d = dst[edge];
    float nrm = dinv[s] * dinv[d];
    const float4* xr = XW4 + (size_t)s * OUT4;
    float* ar = AGG + (size_t)d * OUT4 * 4;
    for (int f = lane; f < OUT4; f += 32) {
        float4 v = xr[f];
        float* a = ar + f * 4;
        atomicAdd(a + 0, v.x * nrm);
        atomicAdd(a + 1, v.y * nrm);
        atomicAdd(a + 2, v.z * nrm);
        atomicAdd(a + 3, v.w * nrm);
    }
}

// h = relu(AGG) + Hres (in place), also emit f16 copy for next GEMM
__global__ void gcn_relu_res_v4(const float4* __restrict__ AGG4, float4* __restrict__ Hres4,
                                v4h* __restrict__ Hh4, int n4) {
    int i = blockIdx.x * blockDim.x + threadIdx.x;
    if (i >= n4) return;
    float4 a = AGG4[i];
    float4 r = Hres4[i];
    float4 v = { fmaxf(a.x, 0.0f) + r.x, fmaxf(a.y, 0.0f) + r.y,
                 fmaxf(a.z, 0.0f) + r.z, fmaxf(a.w, 0.0f) + r.w };
    Hres4[i] = v;
    v4h h = { (_Float16)v.x, (_Float16)v.y, (_Float16)v.z, (_Float16)v.w };
    Hh4[i] = h;
}

// ---------------------------------------------------------------------------
// launch
// ---------------------------------------------------------------------------
extern "C" void kernel_launch(void* const* d_in, const int* in_sizes, int n_in,
                              void* d_out, int out_size, void* d_ws, size_t ws_size,
                              hipStream_t stream) {
    const float* x   = (const float*)d_in[0];
    const int*   ei  = (const int*)d_in[1];       // [2, E]: row 0 = src, row 1 = dst
    const float* W0  = (const float*)d_in[2];
    const float* b0  = (const float*)d_in[3];
    const float* Wp  = (const float*)d_in[4];
    const float* bp  = (const float*)d_in[5];
    const float* W1  = (const float*)d_in[6];
    const float* b1  = (const float*)d_in[7];
    const float* W2  = (const float*)d_in[8];
    const float* b2  = (const float*)d_in[9];
    const float* W3  = (const float*)d_in[10];
    const float* b3  = (const float*)d_in[11];
    float* out = (float*)d_out;

    const int N = GCN_N, E = GCN_E, FIN = GCN_FIN, H = GCN_H, C = GCN_C;
    const int CPAD = 48;
    const int* src = ei;
    const int* dst = ei + E;

    // workspace carve-out (256B aligned)
    char* base = (char*)d_ws;
    size_t off = 0;
    auto carve = [&](size_t bytes) -> void* {
        off = (off + 255) & ~(size_t)255;
        void* p = base + off;
        off += bytes;
        return p;
    };
    float*    dinv = (float*)   carve((size_t)N * sizeof(float));
    _Float16* Xh   = (_Float16*)carve((size_t)N * FIN * sizeof(_Float16));
    _Float16* Hh   = (_Float16*)carve((size_t)N * H   * sizeof(_Float16));
    float*    XW   = (float*)   carve((size_t)N * H   * sizeof(float));
    float*    AGG  = (float*)   carve((size_t)N * H   * sizeof(float));
    float*    Hres = (float*)   carve((size_t)N * H   * sizeof(float));
    _Float16* Wpt  = (_Float16*)carve((size_t)H * FIN * sizeof(_Float16));
    _Float16* W0t  = (_Float16*)carve((size_t)H * FIN * sizeof(_Float16));
    _Float16* W1t  = (_Float16*)carve((size_t)H * H   * sizeof(_Float16));
    _Float16* W2t  = (_Float16*)carve((size_t)H * H   * sizeof(_Float16));
    _Float16* W3t  = (_Float16*)carve((size_t)CPAD * H * sizeof(_Float16));

    const int TPB = 256;
    auto blk = [](long long n, int t) { return (unsigned)((n + t - 1) / t); };

    // --- weight transposes + f16 conversion (tiny, L2-resident afterwards) ---
    gcn_w_transpose<<<blk((long long)FIN * H, TPB), TPB, 0, stream>>>(Wp, Wpt, FIN, H, H);
    gcn_w_transpose<<<blk((long long)FIN * H, TPB), TPB, 0, stream>>>(W0, W0t, FIN, H, H);
    gcn_w_transpose<<<blk((long long)H * H,   TPB), TPB, 0, stream>>>(W1, W1t, H, H, H);
    gcn_w_transpose<<<blk((long long)H * H,   TPB), TPB, 0, stream>>>(W2, W2t, H, H, H);
    gcn_w_transpose<<<blk((long long)H * CPAD,TPB), TPB, 0, stream>>>(W3, W3t, H, C, CPAD);

    // --- symmetric normalization ---
    gcn_deg_init<<<blk(N, TPB), TPB, 0, stream>>>(dinv, N);
    gcn_deg_edges<<<blk(E, TPB), TPB, 0, stream>>>(dst, dinv, E);
    gcn_dinv<<<blk(N, TPB), TPB, 0, stream>>>(dinv, N);

    // --- x -> f16 ---
    gcn_f32_to_f16_v4<<<blk((long long)N * FIN / 4, TPB), TPB, 0, stream>>>(
        (const float4*)x, (v4h*)Xh, N * FIN / 4);

    const unsigned tilesM = N / 16;             // 3125, exact
    const dim3 gemmGridH(tilesM);               // OUT = 128 -> 8 waves/block
    const dim3 gemmBlkH(32 * (H / 16));         // 256 threads
    const dim3 gemmBlkC(32 * (CPAD / 16));      // 96 threads
    const unsigned scatBlocks = blk((long long)E * 32, TPB);
    const int H4 = H / 4, C4 = C / 4;

    // --- identity projection: Hres = x @ Wp + bp ---
    gcn_wmma_gemm<GCN_FIN><<<gemmGridH, gemmBlkH, 0, stream>>>(Xh, Wpt, bp, Hres, H, H);

    // --- layer 0: conv(x, W0) ---
    gcn_wmma_gemm<GCN_FIN><<<gemmGridH, gemmBlkH, 0, stream>>>(Xh, W0t, nullptr, XW, H, H);
    gcn_agg_init<<<blk((long long)N * H4, TPB), TPB, 0, stream>>>(
        (const float4*)XW, dinv, (const float4*)b0, (float4*)AGG, N, H4);
    gcn_edge_scatter<<<scatBlocks, TPB, 0, stream>>>(
        (const float4*)XW, dinv, src, dst, AGG, E, H4);
    gcn_relu_res_v4<<<blk((long long)N * H4, TPB), TPB, 0, stream>>>(
        (const float4*)AGG, (float4*)Hres, (v4h*)Hh, N * H4);

    // --- layers 1, 2: conv(h, Wi) + residual ---
    const float* Wb[2]  = {b1, b2};
    _Float16*    Wt2[2] = {W1t, W2t};
    for (int L = 0; L < 2; ++L) {
        gcn_wmma_gemm<GCN_H><<<gemmGridH, gemmBlkH, 0, stream>>>(Hh, Wt2[L], nullptr, XW, H, H);
        gcn_agg_init<<<blk((long long)N * H4, TPB), TPB, 0, stream>>>(
            (const float4*)XW, dinv, (const float4*)Wb[L], (float4*)AGG, N, H4);
        gcn_edge_scatter<<<scatBlocks, TPB, 0, stream>>>(
            (const float4*)XW, dinv, src, dst, AGG, E, H4);
        gcn_relu_res_v4<<<blk((long long)N * H4, TPB), TPB, 0, stream>>>(
            (const float4*)AGG, (float4*)Hres, (v4h*)Hh, N * H4);
    }

    // --- output layer: conv(h, W3) -> d_out (no relu/residual) ---
    gcn_wmma_gemm<GCN_H><<<gemmGridH, gemmBlkC, 0, stream>>>(Hh, W3t, nullptr, XW, C, C);
    gcn_agg_init<<<blk((long long)N * C4, TPB), TPB, 0, stream>>>(
        (const float4*)XW, dinv, (const float4*)b3, (float4*)out, N, C4);
    gcn_edge_scatter<<<scatBlocks, TPB, 0, stream>>>(
        (const float4*)XW, dinv, src, dst, out, E, C4);
}
